// MultiHeadSelfAttention_34720515621645
// MI455X (gfx1250) — matrix operand order repaired and checked by reading, compiled, not run on hardware
//
#include <hip/hip_runtime.h>

// ---------- types ----------
typedef __attribute__((ext_vector_type(16))) __bf16 v16bf;
typedef __attribute__((ext_vector_type(8)))  __bf16 v8bf;
typedef __attribute__((ext_vector_type(8)))  float  v8f;
typedef __attribute__((ext_vector_type(4)))  unsigned int u32x4;
typedef __attribute__((ext_vector_type(8)))  int i32x8;
typedef __attribute__((ext_vector_type(4)))  int i32x4;

#define BATCH 4
#define TT    2048
#define CC    1024
#define HH    16
#define DD    64
#define MTOT  (BATCH * TT)   // 8192

__device__ inline v16bf load_frag2(const __bf16* p0, const __bf16* p1) {
  union { v16bf v; v8bf h[2]; } u;
  u.h[0] = *(const v8bf*)p0;
  u.h[1] = *(const v8bf*)p1;
  return u.v;
}

__device__ inline v8f wmma_bf16(v16bf a, v16bf b, v8f c) {
  return __builtin_amdgcn_wmma_f32_16x16x32_bf16(
      /*neg_a=*/false, a, /*neg_b=*/false, b,
      /*c_mod=*/(short)0, c, /*reuse_a=*/false, /*reuse_b=*/false);
}

#if __has_builtin(__builtin_amdgcn_tensor_load_to_lds)
#define HAVE_TDM 1
// TDM 2-D tile load: tile_d1 rows of tile_d0 elements (2B each), row stride
// `stride0` elements, from gptr into LDS at lds_off (packed row-major).
// Descriptor layout per CDNA5 ISA.
__device__ inline void tdm_load_tile_2d(unsigned lds_off, const __bf16* gptr,
                                        unsigned tensor_d0, unsigned tensor_d1,
                                        unsigned tile_d0, unsigned tile_d1,
                                        unsigned stride0) {
  unsigned long long ga = (unsigned long long)gptr;
  u32x4 g0;
  g0[0] = 1u;                                   // count=1, user descriptor
  g0[1] = lds_off;                              // LDS byte address
  g0[2] = (unsigned)(ga & 0xffffffffu);         // global_addr[31:0]
  g0[3] = (unsigned)((ga >> 32) & 0x01ffffffu)  // global_addr[56:32]
        | 0x80000000u;                          // type=2 ("image")
  i32x8 g1;
  g1[0] = (int)(1u << 16);                      // data_size=1 -> 2 bytes
  g1[1] = (int)((tensor_d0 & 0xffffu) << 16);                      // dim0[15:0]
  g1[2] = (int)(((tensor_d0 >> 16) & 0xffffu) |
                ((tensor_d1 & 0xffffu) << 16));                    // dim0 hi | dim1 lo
  g1[3] = (int)(((tensor_d1 >> 16) & 0xffffu) | (tile_d0 << 16));  // dim1 hi | tile_dim0
  g1[4] = (int)(tile_d1 & 0xffffu);                                // tile_dim1 (tile_dim2=0)
  g1[5] = (int)stride0;                                            // dim0_stride[31:0]
  g1[6] = 0;                                                       // stride0 hi | stride1 lo
  g1[7] = 0;
  i32x4 z4 = {0, 0, 0, 0};
#if defined(__clang_major__) && __clang_major__ >= 23
  i32x8 z8 = {0, 0, 0, 0, 0, 0, 0, 0};
  __builtin_amdgcn_tensor_load_to_lds(g0, g1, z4, z4, z8, 0);
#else
  __builtin_amdgcn_tensor_load_to_lds(g0, g1, z4, z4, 0);
#endif
}
#else
#define HAVE_TDM 0
#endif

// ---------- fp32 -> bf16 convert ----------
__global__ __launch_bounds__(256)
void conv_f32_bf16(const float* __restrict__ s, __bf16* __restrict__ d, int n) {
  int i = blockIdx.x * blockDim.x + threadIdx.x;
  int stride = gridDim.x * blockDim.x;
  for (; i < n; i += stride) d[i] = (__bf16)s[i];
}

// ---------- tiled WMMA GEMM: Y = X(bf16)[M,K] * W(bf16)[K,N] + bias ----------
// block tile 128(M) x 128(N), K-step 64, double-buffered LDS.
// 256 threads = 8 waves (4x2), wave tile 32x64 -> 16 WMMA per stage.
// A tile staged by TDM (wave 0, overlapped with compute), B staged transposed
// with global loads issued a stage ahead.
template<int N, int K, bool OUTF32>
__global__ __launch_bounds__(256)
void gemm_bf16_kernel(const __bf16* __restrict__ A,   // [M,K]
                      const __bf16* __restrict__ Bw,  // [K,N]
                      const float*  __restrict__ bias,// [N]
                      void* __restrict__ outp,
                      float qscale, int qcols)
{
  __shared__ __align__(16) __bf16 As[2][128][64];  // 2 x 16 KB, row-major [m][k]
  __shared__ __align__(16) __bf16 Bs[2][128][64];  // 2 x 16 KB, transposed [n][k]

  const int tid   = threadIdx.x;
  const int lane  = tid & 31;
  const int wave  = tid >> 5;
  const int laneN = lane & 15;
  const int half  = lane >> 4;
  const int wm    = wave & 3;               // 0..3 -> rows wm*32
  const int wn    = wave >> 2;              // 0..1 -> cols wn*64
  const int m0    = blockIdx.y * 128;
  const int n0    = blockIdx.x * 128;

  v8f zero = {};
  v8f acc[2][4];
#pragma unroll
  for (int i = 0; i < 2; ++i)
#pragma unroll
    for (int j = 0; j < 4; ++j) acc[i][j] = zero;

  // B cooperative-load mapping: 4 threads cover 128 cols of one k-row
  const int bk = tid >> 2;          // 0..63
  const int bn = (tid & 3) * 32;    // 0,32,64,96
#if !HAVE_TDM
  const int arow = tid >> 1;        // 0..127
  const int acol = (tid & 1) * 32;  // 0,32
#endif
  const unsigned as_lds0 = (unsigned)(unsigned long long)&As[0][0][0];
  const unsigned as_lds1 = (unsigned)(unsigned long long)&As[1][0][0];

  // ---- prologue: stage 0 ----
#if HAVE_TDM
  if (wave == 0)
    tdm_load_tile_2d(as_lds0, A + (size_t)m0 * K,
                     /*tensor_d0=*/K, /*tensor_d1=*/128,
                     /*tile_d0=*/64, /*tile_d1=*/128, /*stride0=*/K);
#else
  {
    const __bf16* ga = A + (size_t)(m0 + arow) * K + acol;
#pragma unroll
    for (int s = 0; s < 4; ++s)
      *(v8bf*)&As[0][arow][acol + 8 * s] = *(const v8bf*)(ga + 8 * s);
  }
#endif
  {
    const __bf16* gb = Bw + (size_t)bk * N + n0 + bn;
    v8bf pb[4];
#pragma unroll
    for (int s = 0; s < 4; ++s) pb[s] = *(const v8bf*)(gb + 8 * s);
#pragma unroll
    for (int j = 0; j < 32; ++j) Bs[0][bn + j][bk] = pb[j >> 3][j & 7];
  }
#if HAVE_TDM
  if (wave == 0) __builtin_amdgcn_s_wait_tensorcnt((short)0);
#endif
  __syncthreads();

  // ---- pipelined main loop ----
  for (int k0 = 0; k0 < K; k0 += 64) {
    const int cur = (k0 >> 6) & 1;
    const int nxt = cur ^ 1;
    const bool hasNext = (k0 + 64 < K);

    // issue next stage (TDM A + global B loads) before computing current
    v8bf pb[4];
    if (hasNext) {
#if HAVE_TDM
      if (wave == 0)
        tdm_load_tile_2d(cur ? as_lds0 : as_lds1, A + (size_t)m0 * K + k0 + 64,
                         K, 128, 64, 128, K);
#endif
      const __bf16* gb = Bw + (size_t)(k0 + 64 + bk) * N + n0 + bn;
#pragma unroll
      for (int s = 0; s < 4; ++s) pb[s] = *(const v8bf*)(gb + 8 * s);
    }

    // compute current stage: two 32-deep WMMA chunks
#pragma unroll
    for (int kk = 0; kk < 2; ++kk) {
      const int kof = kk * 32;
      v16bf bfr[4], afr[2];
#pragma unroll
      for (int tj = 0; tj < 4; ++tj) {
        const int nB = wn * 64 + tj * 16 + laneN;
        bfr[tj] = load_frag2(&Bs[cur][nB][kof + half * 16],
                             &Bs[cur][nB][kof + half * 16 + 8]);
      }
#pragma unroll
      for (int ti = 0; ti < 2; ++ti) {
        const int rA = wm * 32 + ti * 16 + laneN;
        afr[ti] = load_frag2(&As[cur][rA][kof + half * 16],
                             &As[cur][rA][kof + half * 16 + 8]);
      }
#pragma unroll
      for (int ti = 0; ti < 2; ++ti)
#pragma unroll
        for (int tj = 0; tj < 4; ++tj)
          acc[ti][tj] = wmma_bf16(afr[ti], bfr[tj], acc[ti][tj]);
    }

    // commit next B tile to LDS (loads had the compute phase to land)
    if (hasNext) {
#pragma unroll
      for (int j = 0; j < 32; ++j) Bs[nxt][bn + j][bk] = pb[j >> 3][j & 7];
#if !HAVE_TDM
      const __bf16* ga = A + (size_t)(m0 + arow) * K + k0 + 64 + acol;
#pragma unroll
      for (int s = 0; s < 4; ++s)
        *(v8bf*)&As[nxt][arow][acol + 8 * s] = *(const v8bf*)(ga + 8 * s);
#endif
    }
#if HAVE_TDM
    if (wave == 0) __builtin_amdgcn_s_wait_tensorcnt((short)0);
#endif
    __syncthreads();
  }

  // epilogue: bias (+ optional per-column scale, used to fold 1/sqrt(D) into Q)
#pragma unroll
  for (int ti = 0; ti < 2; ++ti) {
#pragma unroll
    for (int tj = 0; tj < 4; ++tj) {
      const int col = n0 + wn * 64 + tj * 16 + laneN;
      const float bb = bias[col];
      const float sc = (col < qcols) ? qscale : 1.0f;
#pragma unroll
      for (int r = 0; r < 8; ++r) {
        const int row = m0 + wm * 32 + ti * 16 + half * 8 + r;
        const float v = (acc[ti][tj][r] + bb) * sc;
        if (OUTF32) ((float*)outp)[(size_t)row * N + col]  = v;
        else        ((__bf16*)outp)[(size_t)row * N + col] = (__bf16)v;
      }
    }
  }
}

// ---------- flash attention: 64 query rows per block, 4 waves x 16 rows ----------
// Q is pre-scaled by 1/sqrt(D) in the QKV GEMM epilogue.
__global__ __launch_bounds__(128)
void attn_kernel(const __bf16* __restrict__ qkv,   // [B*T, 3C]
                 __bf16* __restrict__ outb)        // [B*T, C]
{
#if HAVE_TDM
  __shared__ __align__(16) __bf16 Kl[32][64];      // K block (key x d), TDM-staged
#endif
  __shared__ __align__(16) __bf16 Vt[64][32];      // V block transposed (d x key)
  __shared__ __align__(16) __bf16 Pl[4][16][32];   // per-wave P staging

  const int tid   = threadIdx.x;
  const int lane  = tid & 31;
  const int wave  = tid >> 5;
  const int laneN = lane & 15;
  const int half  = lane >> 4;

  const int nqb = TT / 64;
  const int qb  = blockIdx.x % nqb;
  const int h   = (blockIdx.x / nqb) % HH;
  const int b   = blockIdx.x / (nqb * HH);
  const int q0  = qb * 64;
  const int qw0 = q0 + wave * 16;

  const size_t rs = 3 * CC;
  const __bf16* Qb = qkv + (size_t)(b * TT) * rs + h * DD;
  const __bf16* Kb = Qb + CC;
  const __bf16* Vb = Qb + 2 * CC;

  // Q fragments for both 32-wide d-chunks (held in registers for whole loop)
  v16bf qf[2];
  {
    const __bf16* qrow = Qb + (size_t)(qw0 + laneN) * rs;
#pragma unroll
    for (int c = 0; c < 2; ++c)
      qf[c] = load_frag2(qrow + c * 32 + half * 16, qrow + c * 32 + half * 16 + 8);
  }

  v8f zero = {};
  v8f o[4];
#pragma unroll
  for (int j = 0; j < 4; ++j) o[j] = zero;
  float row_m[8], row_l[8];
#pragma unroll
  for (int r = 0; r < 8; ++r) { row_m[r] = -1e30f; row_l[r] = 0.0f; }

  const int vk = tid >> 2;          // 0..31 (key within block)
  const int vd = (tid & 3) * 16;    // 0,16,32,48 (d segment)
  const size_t kstep = 32 * rs;

  const __bf16* vrow  = Vb + (size_t)vk * rs + vd;
#if HAVE_TDM
  const __bf16* ktile = Kb;
  const unsigned kl_lds = (unsigned)(unsigned long long)&Kl[0][0];
#else
  const __bf16* kp0 = Kb + (size_t)laneN * rs + half * 16;        // tile 0
  const __bf16* kp1 = Kb + (size_t)(16 + laneN) * rs + half * 16; // tile 1
#endif

  const int kend = q0 + 64;
  for (int kb = 0; kb < kend; kb += 32) {
    __syncthreads();
#if HAVE_TDM
    if (wave == 0)   // stage K block (32 keys x 64 d) via tensor DMA
      tdm_load_tile_2d(kl_lds, ktile, /*tensor_d0=*/64, /*tensor_d1=*/32,
                       /*tile_d0=*/64, /*tile_d1=*/32, /*stride0=*/(unsigned)rs);
#endif
    { // stage V block transposed into LDS
      v8bf v0 = *(const v8bf*)vrow;
      v8bf v1 = *(const v8bf*)(vrow + 8);
#pragma unroll
      for (int j = 0; j < 8; ++j) { Vt[vd + j][vk] = v0[j]; Vt[vd + 8 + j][vk] = v1[j]; }
      vrow += kstep;
    }
#if HAVE_TDM
    if (wave == 0) __builtin_amdgcn_s_wait_tensorcnt((short)0);
    ktile += kstep;
#endif
    __syncthreads();

    if (kb + 32 < kend)  // prefetch next V block
      __builtin_prefetch(vrow, 0, 1);

    // S = Q * K^T for 32 keys: hoist all fragment loads, then 4 WMMAs
#if HAVE_TDM
    const __bf16* kr0 = &Kl[laneN][half * 16];
    const __bf16* kr1 = &Kl[16 + laneN][half * 16];
#else
    const __bf16* kr0 = kp0;
    const __bf16* kr1 = kp1;
    kp0 += kstep; kp1 += kstep;
#endif
    v16bf kf00 = load_frag2(kr0,      kr0 + 8);
    v16bf kf10 = load_frag2(kr1,      kr1 + 8);
    v16bf kf01 = load_frag2(kr0 + 32, kr0 + 40);
    v16bf kf11 = load_frag2(kr1 + 32, kr1 + 40);
    v8f s0 = wmma_bf16(qf[0], kf00, zero);
    v8f s1 = wmma_bf16(qf[0], kf10, zero);
    s0 = wmma_bf16(qf[1], kf01, s0);
    s1 = wmma_bf16(qf[1], kf11, s1);

    // causal mask (wave-uniform skip) + online softmax
    const bool needMask = (kb + 31 > qw0);
#pragma unroll
    for (int r = 0; r < 8; ++r) {
      float a0 = s0[r];
      float a1 = s1[r];
      if (needMask) {
        const int qrow = qw0 + half * 8 + r;
        if (kb + laneN      > qrow) a0 = -1e30f;
        if (kb + 16 + laneN > qrow) a1 = -1e30f;
      }
      float vmax = fmaxf(a0, a1);
#pragma unroll
      for (int off = 1; off < 16; off <<= 1)
        vmax = fmaxf(vmax, __shfl_xor(vmax, off, 32));
      const float mnew = fmaxf(row_m[r], vmax);
      const float corr = __expf(row_m[r] - mnew);
      row_m[r] = mnew;
      const float p0 = __expf(a0 - mnew);
      const float p1 = __expf(a1 - mnew);
      float ps = p0 + p1;
#pragma unroll
      for (int off = 1; off < 16; off <<= 1)
        ps += __shfl_xor(ps, off, 32);
      row_l[r] = row_l[r] * corr + ps;
#pragma unroll
      for (int j = 0; j < 4; ++j) o[j][r] *= corr;
      Pl[wave][half * 8 + r][laneN]      = (__bf16)p0;
      Pl[wave][half * 8 + r][16 + laneN] = (__bf16)p1;
    }

    // O += P * V  (hoist fragment loads, then 4 WMMAs)
    v16bf pf = load_frag2(&Pl[wave][laneN][half * 16], &Pl[wave][laneN][half * 16 + 8]);
    v16bf vf[4];
#pragma unroll
    for (int j = 0; j < 4; ++j) {
      const int dc = j * 16 + laneN;
      vf[j] = load_frag2(&Vt[dc][half * 16], &Vt[dc][half * 16 + 8]);
    }
#pragma unroll
    for (int j = 0; j < 4; ++j)
      o[j] = wmma_bf16(pf, vf[j], o[j]);
  }

  // normalize + store (bf16 attention output, [B*T, C] with head at h*64)
#pragma unroll
  for (int r = 0; r < 8; ++r) {
    const float inv = 1.0f / row_l[r];
    const int row = b * TT + qw0 + half * 8 + r;
#pragma unroll
    for (int j = 0; j < 4; ++j)
      outb[(size_t)row * CC + h * DD + j * 16 + laneN] = (__bf16)(o[j][r] * inv);
  }
}

// ---------- host launcher ----------
extern "C" void kernel_launch(void* const* d_in, const int* in_sizes, int n_in,
                              void* d_out, int out_size, void* d_ws, size_t ws_size,
                              hipStream_t stream) {
  (void)in_sizes; (void)n_in; (void)out_size; (void)ws_size;
  const float* x      = (const float*)d_in[0];
  const float* w_attn = (const float*)d_in[1];
  const float* b_attn = (const float*)d_in[2];
  const float* w_proj = (const float*)d_in[3];
  const float* b_proj = (const float*)d_in[4];
  float* out = (float*)d_out;

  // workspace layout (bf16), all 16B aligned
  char* ws = (char*)d_ws;
  size_t off = 0;
  __bf16* xb    = (__bf16*)(ws + off); off += (size_t)MTOT * CC * 2;       // 16 MB
  __bf16* wqkvb = (__bf16*)(ws + off); off += (size_t)CC * 3 * CC * 2;     //  6 MB
  __bf16* wprjb = (__bf16*)(ws + off); off += (size_t)CC * CC * 2;         //  2 MB
  __bf16* qkvb  = (__bf16*)(ws + off); off += (size_t)MTOT * 3 * CC * 2;   // 48 MB
  __bf16* attnb = (__bf16*)(ws + off); off += (size_t)MTOT * CC * 2;       // 16 MB

  const int n_x  = MTOT * CC;
  const int n_wa = CC * 3 * CC;
  const int n_wp = CC * CC;
  conv_f32_bf16<<<(n_x  + 255) / 256, 256, 0, stream>>>(x,      xb,    n_x);
  conv_f32_bf16<<<(n_wa + 255) / 256, 256, 0, stream>>>(w_attn, wqkvb, n_wa);
  conv_f32_bf16<<<(n_wp + 255) / 256, 256, 0, stream>>>(w_proj, wprjb, n_wp);

  // QKV projection: [8192,1024] x [1024,3072] -> bf16 qkv, Q cols pre-scaled
  gemm_bf16_kernel<3 * CC, CC, false>
      <<<dim3(3 * CC / 128, MTOT / 128), 256, 0, stream>>>(
          xb, wqkvb, b_attn, qkvb, 0.125f, CC);

  // attention: B*H*(T/64) blocks of 128 threads
  attn_kernel<<<dim3(BATCH * HH * (TT / 64)), 128, 0, stream>>>(qkvb, attnb);

  // output projection: [8192,1024] x [1024,1024] -> fp32 out
  gemm_bf16_kernel<CC, CC, true>
      <<<dim3(CC / 128, MTOT / 128), 256, 0, stream>>>(
          attnb, wprjb, b_proj, out, 1.0f, 0);
}
